// BertFreezeSegmentor_15290083574129
// MI455X (gfx1250) — compile-verified, matrix operand order checked by reading, not run on hardware
//
#include <hip/hip_runtime.h>

// ---------------------------------------------------------------------------
// CDNA5 (gfx1250): BiLSTM + stack-decode.
// All matmuls via v_wmma_f32_16x16x32_bf16 (bf16 in, f32 accumulate).
// Sequential phases run as persistent cooperative kernels (24 blocks) with a
// device-scope generation barrier instead of per-step kernel launches.
// ---------------------------------------------------------------------------

typedef __attribute__((ext_vector_type(16))) __bf16 v16bf;
typedef __attribute__((ext_vector_type(8)))  float  v8f;

union Frag {
    v16bf v;
    uint4 q[2];
};

__device__ __forceinline__ float sigm(float x) { return 1.0f / (1.0f + expf(-x)); }

// ---------------------------------------------------------------------------
// Device-wide generation barrier (deterministic; 24 co-resident blocks).
// bar[0] = arrival counter, bar[1] = generation.
// ---------------------------------------------------------------------------
__device__ __forceinline__ void gsync(unsigned* bar, unsigned nb)
{
    __syncthreads();
    if (threadIdx.x == 0) {
        __threadfence();
        unsigned gen = __hip_atomic_load(&bar[1], __ATOMIC_RELAXED, __HIP_MEMORY_SCOPE_AGENT);
        unsigned arrived = __hip_atomic_fetch_add(&bar[0], 1u, __ATOMIC_ACQ_REL, __HIP_MEMORY_SCOPE_AGENT);
        if (arrived == nb - 1) {
            __hip_atomic_store(&bar[0], 0u, __ATOMIC_RELAXED, __HIP_MEMORY_SCOPE_AGENT);
            __hip_atomic_fetch_add(&bar[1], 1u, __ATOMIC_RELEASE, __HIP_MEMORY_SCOPE_AGENT);
        } else {
            while (__hip_atomic_load(&bar[1], __ATOMIC_ACQUIRE, __HIP_MEMORY_SCOPE_AGENT) == gen)
                __builtin_amdgcn_s_sleep(1);
        }
    }
    __syncthreads();
}

// ---------------------------------------------------------------------------
// WMMA fragment helpers. A lane layout (ISA 7.12.2, 16-bit A 16x32): lane L ->
// row (L&15), K-halves at (L>>4)*8 and (L>>4)*8+16. B mirrors A; weight rows
// of W(N,K) are exactly B columns (contiguous). ap/wp are pre-offset per lane.
// ---------------------------------------------------------------------------
template<int KROW>
__device__ __forceinline__ void load_frags(const __bf16* __restrict__ ap,
                                           const __bf16* __restrict__ wp,
                                           int kk, Frag& a, Frag b[4])
{
    a.q[0] = *(const uint4*)(ap + kk);
    a.q[1] = *(const uint4*)(ap + kk + 16);
#pragma unroll
    for (int tn = 0; tn < 4; ++tn) {
        const __bf16* wr = wp + (size_t)(tn * 16) * KROW + kk;
        b[tn].q[0] = *(const uint4*)(wr);
        b[tn].q[1] = *(const uint4*)(wr + 16);
    }
}

__device__ __forceinline__ void wmma4(const Frag& a, const Frag b[4], v8f acc[4])
{
#pragma unroll
    for (int tn = 0; tn < 4; ++tn)
        acc[tn] = __builtin_amdgcn_wmma_f32_16x16x32_bf16(
            false, a.v, false, b[tn].v, (short)0, acc[tn], false, false);
}

// Ping-pong pipelined inner product for one 16x64 strip (4 accumulators).
// No inter-buffer copies. sched_group_barrier directives force the scheduler
// to interleave next-step VMEM reads with current-step WMMAs (software
// pipelining) instead of clustering load -> wait(0) -> wmma.
template<int K>
__device__ __forceinline__ void gemm_core(const __bf16* __restrict__ ap,
                                          const __bf16* __restrict__ wp,
                                          v8f acc[4])
{
    static_assert(K % 64 == 0, "K must be a multiple of 64");
    Frag aA, aB;
    Frag bA[4], bB[4];
    load_frags<K>(ap, wp, 0, aA, bA);
#pragma unroll
    for (int kk = 0; kk < K; kk += 64) {
        load_frags<K>(ap, wp, kk + 32, aB, bB);
        wmma4(aA, bA, acc);
        if (kk + 64 < K)
            load_frags<K>(ap, wp, kk + 64, aA, bA);
        wmma4(aB, bB, acc);
        // Interleave: 3 VMEM reads (0x20) : 1 matrix op (0x08), 8x per step
        // (20 loads + 8 WMMAs per 64-wide K step).
#pragma unroll
        for (int i = 0; i < 8; ++i) {
            __builtin_amdgcn_sched_group_barrier(0x0020, 3, 0);
            __builtin_amdgcn_sched_group_barrier(0x0008, 1, 0);
        }
    }
}

__device__ __forceinline__ void acc_bias(v8f acc[4], const float* __restrict__ bias,
                                         int n0, int lid)
{
#pragma unroll
    for (int tn = 0; tn < 4; ++tn) {
        const float bv = bias[n0 + (tn << 4) + lid];
#pragma unroll
        for (int v = 0; v < 8; ++v) acc[tn][v] = bv;
    }
}

// D layout (32-bit C/D 16x16): lane L -> col (L&15); VGPR v -> row v + 8*(L>>4)
__device__ __forceinline__ void store_f32(float* __restrict__ C, int N, int m0, int n0,
                                          int lid, int half, const v8f acc[4])
{
#pragma unroll
    for (int tn = 0; tn < 4; ++tn) {
        const int n = n0 + (tn << 4) + lid;
#pragma unroll
        for (int v = 0; v < 8; ++v)
            C[(size_t)(m0 + v + (half << 3)) * N + n] = acc[tn][v];
    }
}

// ---------------------------------------------------------------------------
// Parallel input projection: Cbf(M,3072) = A(M,768) @ W(3072,768)^T + bias
// ---------------------------------------------------------------------------
__global__ __launch_bounds__(256)
void proj_gemm(const __bf16* __restrict__ A, const __bf16* __restrict__ W,
               const float* __restrict__ bias, __bf16* __restrict__ Cbf, int M)
{
    const int lane = threadIdx.x & 31;
    const int wid  = (int)((blockIdx.x * 256u + threadIdx.x) >> 5);
    const int total = (M >> 4) * 48;
    if (wid >= total) return;                       // wave-uniform
    const int m0 = (wid / 48) << 4;
    const int n0 = (wid % 48) << 6;
    const int half = lane >> 4, lid = lane & 15;

    const __bf16* ap = A + (size_t)(m0 + lid) * 768 + (half << 3);
    const __bf16* wp = W + (size_t)(n0 + lid) * 768 + (half << 3);
    v8f acc[4];
    acc_bias(acc, bias, n0, lid);
    gemm_core<768>(ap, wp, acc);
#pragma unroll
    for (int tn = 0; tn < 4; ++tn) {
        const int n = n0 + (tn << 4) + lid;
#pragma unroll
        for (int v = 0; v < 8; ++v)
            Cbf[(size_t)(m0 + v + (half << 3)) * 3072 + n] = (__bf16)acc[tn][v];
    }
}

__global__ __launch_bounds__(256)
void f32_to_bf16(const float* __restrict__ src, __bf16* __restrict__ dst, size_t n)
{
    size_t i = (size_t)blockIdx.x * blockDim.x + threadIdx.x;
    if (i < n) dst[i] = (__bf16)src[i];
}

// ---------------------------------------------------------------------------
// Persistent BiLSTM recurrence (one direction). Grid = 24 x 256 = 192 waves.
// Per step: [GEMM h(t-prev) @ Whh^T -> hwhh] | barrier | [gates] | barrier
// hwhh & cstate pre-zeroed (step 0 uses them as zeros).
// ---------------------------------------------------------------------------
__global__ __launch_bounds__(256)
void lstm_recurrence(const __bf16* __restrict__ G,     // (B*S,3072) = x@Wih^T + b
                     const __bf16* __restrict__ Whh,   // (3072,768)
                     __bf16* __restrict__ lstm_bf,     // (B,S,1536)
                     float* __restrict__ hwhh,         // (64,3072)
                     float* __restrict__ cstate,       // (64,768)
                     unsigned* __restrict__ bar,
                     int S, int dir)
{
    const int tid  = blockIdx.x * 256 + threadIdx.x;   // 0..6143
    const int lane = threadIdx.x & 31;
    const int wid  = tid >> 5;                         // 0..191
    const int half = lane >> 4, lid = lane & 15;
    const int m0 = (wid / 48) << 4;
    const int n0 = (wid % 48) << 6;
    const int hoff = dir ? 768 : 0;
    const unsigned NB = gridDim.x;

    const __bf16* wp = Whh + (size_t)(n0 + lid) * 768 + (half << 3);
    const size_t a_row = (size_t)(m0 + lid) * ((size_t)S * 1536);

    for (int step = 0; step < S; ++step) {
        const int t = dir ? (S - 1 - step) : step;
        if (step > 0) {
            const int tprev = dir ? (t + 1) : (t - 1);
            const __bf16* ap = lstm_bf + a_row + (size_t)tprev * 1536 + hoff + (half << 3);
            v8f acc[4];
#pragma unroll
            for (int tn = 0; tn < 4; ++tn)
#pragma unroll
                for (int v = 0; v < 8; ++v) acc[tn][v] = 0.0f;
            gemm_core<768>(ap, wp, acc);
            store_f32(hwhh, 3072, m0, n0, lid, half, acc);
        }
        gsync(bar, NB);

        for (int e = tid; e < 64 * 768; e += 6144) {
            const int b = e / 768, j = e - b * 768;
            const size_t gb = ((size_t)b * S + t) * 3072;
            const int hb = b * 3072;
            const float gi = hwhh[hb + j]        + (float)G[gb + j];
            const float gf = hwhh[hb + 768 + j]  + (float)G[gb + 768 + j];
            const float gg = hwhh[hb + 1536 + j] + (float)G[gb + 1536 + j];
            const float go = hwhh[hb + 2304 + j] + (float)G[gb + 2304 + j];
            const float c  = sigm(gf) * cstate[e] + sigm(gi) * tanhf(gg);
            cstate[e] = c;
            lstm_bf[((size_t)b * S + t) * 1536 + hoff + j] = (__bf16)(sigm(go) * tanhf(c));
        }
        gsync(bar, NB);
    }
}

// ---------------------------------------------------------------------------
// Persistent decode. Grid = 24 x 256 = 192 waves. Per step 5 phases:
//  1) g1 = x_prev@sWih^T + sb + sh[b,sp]@sWhh^T      (fused two-pass GEMM)
//  2) sub-word gates -> sh/sc at sp+1, subword=[h1,c1]
//  3) g2 = subword@wWih^T + wb + wh[b,wp]@wWhh^T
//  4) word gates -> wh/wc at wp+1, wh1
//  5) classifier (wave-shuffle reduction) + sp/wp update
// ---------------------------------------------------------------------------
__global__ __launch_bounds__(256)
void decode_persistent(const __bf16* __restrict__ lstm_bf,
                       const __bf16* __restrict__ sWih, const __bf16* __restrict__ sWhh,
                       const float* __restrict__ sbias,
                       const __bf16* __restrict__ wWih, const __bf16* __restrict__ wWhh,
                       const float* __restrict__ wbias,
                       const float* __restrict__ clsW, const float* __restrict__ clsb,
                       const int* __restrict__ golds,
                       __bf16* __restrict__ sh, float* __restrict__ sc,
                       __bf16* __restrict__ wh, float* __restrict__ wc,
                       __bf16* __restrict__ subword, float* __restrict__ wh1,
                       float* __restrict__ g1, float* __restrict__ g2,
                       int* __restrict__ sp, int* __restrict__ wp,
                       float* __restrict__ out,
                       unsigned* __restrict__ bar, int S)
{
    const int tid  = blockIdx.x * 256 + threadIdx.x;
    const int lane = threadIdx.x & 31;
    const int wid  = tid >> 5;
    const int half = lane >> 4, lid = lane & 15;
    const int m0 = (wid / 48) << 4;
    const int n0 = (wid % 48) << 6;
    const unsigned NB = gridDim.x;
    const int arow = m0 + lid;

    const __bf16* swih_p = sWih + (size_t)(n0 + lid) * 1536 + (half << 3);
    const __bf16* swhh_p = sWhh + (size_t)(n0 + lid) * 768  + (half << 3);
    const __bf16* wwih_p = wWih + (size_t)(n0 + lid) * 1536 + (half << 3);
    const __bf16* wwhh_p = wWhh + (size_t)(n0 + lid) * 768  + (half << 3);

    for (int t = 0; t < S - 1; ++t) {
        // ---- phase 1: sub-word GEMM -> g1 ---------------------------------
        {
            v8f acc[4];
            acc_bias(acc, sbias, n0, lid);
            const __bf16* ap1 = lstm_bf + ((size_t)arow * S + t) * 1536 + (half << 3);
            gemm_core<1536>(ap1, swih_p, acc);
            const int p = sp[arow];
            const __bf16* ap2 = sh + ((size_t)arow * 514 + p) * 768 + (half << 3);
            gemm_core<768>(ap2, swhh_p, acc);
            store_f32(g1, 3072, m0, n0, lid, half, acc);
        }
        gsync(bar, NB);

        // ---- phase 2: sub-word gates --------------------------------------
        for (int e = tid; e < 64 * 768; e += 6144) {
            const int b = e / 768, j = e - b * 768;
            const int p = sp[b];
            const size_t r = ((size_t)b * 514 + p) * 768 + j;
            const size_t w = r + 768;
            const int gb = b * 3072;
            const float gi = g1[gb + j],        gf = g1[gb + 768 + j];
            const float gg = g1[gb + 1536 + j], go = g1[gb + 2304 + j];
            const float c  = sigm(gf) * sc[r] + sigm(gi) * tanhf(gg);
            const float h  = sigm(go) * tanhf(c);
            sc[w] = c;
            sh[w] = (__bf16)h;
            subword[b * 1536 + j]       = (__bf16)h;
            subword[b * 1536 + 768 + j] = (__bf16)c;
        }
        gsync(bar, NB);

        // ---- phase 3: word GEMM -> g2 -------------------------------------
        {
            v8f acc[4];
            acc_bias(acc, wbias, n0, lid);
            const __bf16* ap1 = subword + (size_t)arow * 1536 + (half << 3);
            gemm_core<1536>(ap1, wwih_p, acc);
            const int p = wp[arow];
            const __bf16* ap2 = wh + ((size_t)arow * 257 + p) * 768 + (half << 3);
            gemm_core<768>(ap2, wwhh_p, acc);
            store_f32(g2, 3072, m0, n0, lid, half, acc);
        }
        gsync(bar, NB);

        // ---- phase 4: word gates ------------------------------------------
        for (int e = tid; e < 64 * 768; e += 6144) {
            const int b = e / 768, j = e - b * 768;
            const int p = wp[b];
            const size_t r = ((size_t)b * 257 + p) * 768 + j;
            const size_t w = r + 768;
            const int gb = b * 3072;
            const float gi = g2[gb + j],        gf = g2[gb + 768 + j];
            const float gg = g2[gb + 1536 + j], go = g2[gb + 2304 + j];
            const float c  = sigm(gf) * wc[r] + sigm(gi) * tanhf(gg);
            const float h  = sigm(go) * tanhf(c);
            wc[w] = c;
            wh[w] = (__bf16)h;
            wh1[b * 768 + j] = h;
        }
        gsync(bar, NB);

        // ---- phase 5: classifier + pointer update -------------------------
        if (wid < 128) {
            const int b = wid >> 1, cls = wid & 1;
            float s = 0.0f;
            for (int j = lane; j < 2304; j += 32) {
                const float v = (j < 768)
                    ? wh1[b * 768 + j]
                    : (float)lstm_bf[((size_t)b * S + t + 1) * 1536 + (j - 768)];
                s += v * clsW[cls * 2304 + j];
            }
#pragma unroll
            for (int o = 16; o > 0; o >>= 1) s += __shfl_xor(s, o, 32);
            if (lane == 0) {
                out[((size_t)b * S + t + 1) * 2 + cls] = s + clsb[cls];
                if (cls == 0) {
                    const int g = golds[b * S + t + 1];
                    sp[b] += (g == 0) ? 1 : 2;   // subw_map = [1,2,2]
                    wp[b] += (g == 0) ? 0 : 1;   // word_map = [0,1,1]
                }
            }
        }
        gsync(bar, NB);
    }
}

__global__ void init_out_first(float* __restrict__ out, int S)
{
    const int b = threadIdx.x;   // 64 threads
    out[(size_t)b * S * 2 + 0] = -1.0f;
    out[(size_t)b * S * 2 + 1] =  1.0f;
}

// ---------------------------------------------------------------------------
// host side
// ---------------------------------------------------------------------------
static inline void launch_cvt(hipStream_t s, const float* src, __bf16* dst, size_t n)
{
    f32_to_bf16<<<(unsigned)((n + 255) / 256), 256, 0, s>>>(src, dst, n);
}

extern "C" void kernel_launch(void* const* d_in, const int* in_sizes, int n_in,
                              void* d_out, int out_size, void* d_ws, size_t ws_size,
                              hipStream_t stream)
{
    (void)in_sizes; (void)n_in; (void)out_size; (void)ws_size;

    const float* hidden = (const float*)d_in[0];
    const int*   golds  = (const int*)d_in[1];
    const float* Wih_f  = (const float*)d_in[2];
    const float* Whh_f  = (const float*)d_in[3];
    const float* b_f    = (const float*)d_in[4];
    const float* Wih_b  = (const float*)d_in[5];
    const float* Whh_b  = (const float*)d_in[6];
    const float* b_b    = (const float*)d_in[7];
    const float* sWih   = (const float*)d_in[8];
    const float* sWhh   = (const float*)d_in[9];
    const float* sb     = (const float*)d_in[10];
    const float* wWih   = (const float*)d_in[11];
    const float* wWhh   = (const float*)d_in[12];
    const float* wb     = (const float*)d_in[13];
    const float* clsW   = (const float*)d_in[14];
    const float* clsb   = (const float*)d_in[15];
    float* out = (float*)d_out;

    constexpr int BB = 64, S = 256, H = 768;
    constexpr int G4 = 4 * H;           // 3072
    constexpr int D2 = 2 * H;           // 1536

    char* ws = (char*)d_ws;
    size_t off = 0;
    auto take = [&](size_t bytes) -> char* {
        char* p = ws + off;
        off += (bytes + 255) & ~(size_t)255;
        return p;
    };

    __bf16* wih_f_bf = (__bf16*)take((size_t)G4 * H * 2);
    __bf16* whh_f_bf = (__bf16*)take((size_t)G4 * H * 2);
    __bf16* wih_b_bf = (__bf16*)take((size_t)G4 * H * 2);
    __bf16* whh_b_bf = (__bf16*)take((size_t)G4 * H * 2);
    __bf16* swih_bf  = (__bf16*)take((size_t)G4 * D2 * 2);
    __bf16* swhh_bf  = (__bf16*)take((size_t)G4 * H * 2);
    __bf16* wwih_bf  = (__bf16*)take((size_t)G4 * D2 * 2);
    __bf16* wwhh_bf  = (__bf16*)take((size_t)G4 * H * 2);
    __bf16* X_bf     = (__bf16*)take((size_t)BB * S * H * 2);
    __bf16* lstm_bf  = (__bf16*)take((size_t)BB * S * D2 * 2);   // (B,S,1536)
    float*  hwhh     = (float*) take((size_t)BB * G4 * 4);
    float*  cstate   = (float*) take((size_t)BB * H * 4);
    float*  g1       = (float*) take((size_t)BB * G4 * 4);
    float*  g2       = (float*) take((size_t)BB * G4 * 4);
    __bf16* subword  = (__bf16*)take((size_t)BB * D2 * 2);
    float*  wh1      = (float*) take((size_t)BB * H * 4);
    int*    sp       = (int*)   take((size_t)BB * 4);
    int*    wp       = (int*)   take((size_t)BB * 4);
    unsigned* bar    = (unsigned*)take(256);

    // phase-disjoint arena: {Gf,Gb} (BiLSTM) overlaps decode stacks
    char* arena = ws + off;
    __bf16* Gf_bf = (__bf16*)(arena);
    __bf16* Gb_bf = (__bf16*)(arena + (size_t)BB * S * G4 * 2);
    __bf16* sh_bf = (__bf16*)(arena);
    float*  sc_f  = (float*)(arena + (size_t)BB * 514 * H * 2);
    __bf16* wh_bf = (__bf16*)((char*)sc_f + (size_t)BB * 514 * H * 4);
    float*  wc_f  = (float*)((char*)wh_bf + (size_t)BB * 257 * H * 2);

    // ---- 1. convert weights / inputs to bf16 ------------------------------
    launch_cvt(stream, Wih_f, wih_f_bf, (size_t)G4 * H);
    launch_cvt(stream, Whh_f, whh_f_bf, (size_t)G4 * H);
    launch_cvt(stream, Wih_b, wih_b_bf, (size_t)G4 * H);
    launch_cvt(stream, Whh_b, whh_b_bf, (size_t)G4 * H);
    launch_cvt(stream, sWih,  swih_bf,  (size_t)G4 * D2);
    launch_cvt(stream, sWhh,  swhh_bf,  (size_t)G4 * H);
    launch_cvt(stream, wWih,  wwih_bf,  (size_t)G4 * D2);
    launch_cvt(stream, wWhh,  wwhh_bf,  (size_t)G4 * H);
    launch_cvt(stream, hidden, X_bf,    (size_t)BB * S * H);

    // ---- 2. input projections (parallel WMMA GEMMs) -----------------------
    {
        const int M = BB * S;                       // 16384
        const int grid = (M / 16) * 48 * 32 / 256;  // 6144 blocks
        proj_gemm<<<grid, 256, 0, stream>>>(X_bf, wih_f_bf, b_f, Gf_bf, M);
        proj_gemm<<<grid, 256, 0, stream>>>(X_bf, wih_b_bf, b_b, Gb_bf, M);
    }

    // ---- 3. forward + backward recurrences (persistent) -------------------
    hipMemsetAsync(bar, 0, 256, stream);
    hipMemsetAsync(hwhh,   0, (size_t)BB * G4 * 4, stream);
    hipMemsetAsync(cstate, 0, (size_t)BB * H * 4, stream);
    lstm_recurrence<<<24, 256, 0, stream>>>(Gf_bf, whh_f_bf, lstm_bf,
                                            hwhh, cstate, bar, S, 0);
    hipMemsetAsync(hwhh,   0, (size_t)BB * G4 * 4, stream);
    hipMemsetAsync(cstate, 0, (size_t)BB * H * 4, stream);
    lstm_recurrence<<<24, 256, 0, stream>>>(Gb_bf, whh_b_bf, lstm_bf,
                                            hwhh, cstate, bar, S, 1);

    // ---- 4. decode (persistent) -------------------------------------------
    hipMemsetAsync(sh_bf, 0, (size_t)BB * 514 * H * 2, stream);
    hipMemsetAsync(sc_f,  0, (size_t)BB * 514 * H * 4, stream);
    hipMemsetAsync(wh_bf, 0, (size_t)BB * 257 * H * 2, stream);
    hipMemsetAsync(wc_f,  0, (size_t)BB * 257 * H * 4, stream);
    hipMemsetAsync(sp, 0, (size_t)BB * 4, stream);
    hipMemsetAsync(wp, 0, (size_t)BB * 4, stream);
    init_out_first<<<1, BB, 0, stream>>>(out, S);

    decode_persistent<<<24, 256, 0, stream>>>(lstm_bf,
                                              swih_bf, swhh_bf, sb,
                                              wwih_bf, wwhh_bf, wb,
                                              clsW, clsb, golds,
                                              sh_bf, sc_f, wh_bf, wc_f,
                                              subword, wh1, g1, g2,
                                              sp, wp, out, bar, S);
}
